// GraphTaskNodeGCN2_3659312136457
// MI455X (gfx1250) — matrix operand order repaired and checked by reading, not compiled, not run on hardware
//
#include <hip/hip_runtime.h>
#include <hip/hip_bf16.h>
#include <stdint.h>

// ---------------------------------------------------------------------------
// 2-layer GCN + mean-pool + linear readout for MI455X (gfx1250, wave32).
//  - Dense node-feature GEMMs: v_wmma_f32_16x16x32_bf16, one wave per 16-row
//    strip, A k-tiles resident in VGPRs, B from a pre-transposed bf16 weight
//    copy so every operand load is a b128.
//  - Edge scatter/gather: one wave per edge, float4 gather + f32 atomics
//    (bandwidth-bound phase, ~3.3 GB HBM traffic -> ~150us at 23.3 TB/s).
// ---------------------------------------------------------------------------

typedef __bf16 bf16_t;
typedef bf16_t v16bf __attribute__((ext_vector_type(16)));
typedef float  v8f   __attribute__((ext_vector_type(8)));

// ---------------- utility kernels -------------------------------------------

__global__ void fill4_kernel(float4* __restrict__ p, size_t n4, float v) {
    size_t i = (size_t)blockIdx.x * blockDim.x + threadIdx.x;
    if (i < n4) p[i] = make_float4(v, v, v, v);
}

__global__ void deg_kernel(const int* __restrict__ dst, float* __restrict__ deg, int E) {
    int e = blockIdx.x * blockDim.x + threadIdx.x;
    if (e < E) atomicAdd(&deg[dst[e]], 1.0f);
}

__global__ void dinv_kernel(const float* __restrict__ deg, float* __restrict__ dinv, int n) {
    int i = blockIdx.x * blockDim.x + threadIdx.x;
    if (i < n) {
        float d = deg[i];
        dinv[i] = (d > 0.0f) ? rsqrtf(d) : 0.0f;
    }
}

// Convert + transpose weights: W[K][NC] f32 -> Wt[NC][K] bf16 (runs once, tiny).
__global__ void wcvt_kernel(const float* __restrict__ W, bf16_t* __restrict__ Wt,
                            int K, int NC) {
    int i = blockIdx.x * blockDim.x + threadIdx.x;
    if (i >= K * NC) return;
    int k = i / NC, nc = i % NC;
    Wt[(size_t)nc * K + k] = (bf16_t)W[i];
}

// ---------------- WMMA GEMM: Out[nrows,NC] = A[nrows,K] @ W[K,NC] -----------
// Wt is the bf16 [NC][K] transposed weight copy. One wave per 16-row strip;
// nrows % 16 == 0 (N = 100000 = 6250*16); the tile guard is wave-uniform so
// EXEC is all-ones inside every WMMA.

template <int K, int NC>
__global__ __launch_bounds__(256) void gemm_bf16_wmma(const float* __restrict__ A,
                                                      const bf16_t* __restrict__ Wt,
                                                      float* __restrict__ Out,
                                                      int nrows) {
    constexpr int KT = K / 32;   // k-tiles of 32
    constexpr int NT = NC / 16;  // column tiles of 16

    const int wave = threadIdx.x >> 5;
    const int lane = threadIdx.x & 31;
    const int tile = blockIdx.x * 8 + wave;
    if (tile * 16 >= nrows) return;             // wave-uniform

    const int m0   = tile << 4;
    const int lo   = lane & 15;
    const int half = lane >> 4;

    // A 16x32 bf16 layout (ISA 7.12.2): lane holds row m = lo;
    //   element j   -> K = kt*32 + 8*half + j        (j = 0..7)
    //   element 8+j -> K = kt*32 + 16 + 8*half + j
    // Per k-tile that is two pairs of 16B-contiguous chunks -> 4x b128 loads.
    v16bf a[KT];
    const float* arow = A + (size_t)(m0 + lo) * K + 8 * half;
#pragma unroll
    for (int kt = 0; kt < KT; ++kt) {
        float4 f0 = *(const float4*)(arow + kt * 32);
        float4 f1 = *(const float4*)(arow + kt * 32 + 4);
        float4 f2 = *(const float4*)(arow + kt * 32 + 16);
        float4 f3 = *(const float4*)(arow + kt * 32 + 20);
        a[kt][0]  = (bf16_t)f0.x;  a[kt][1]  = (bf16_t)f0.y;
        a[kt][2]  = (bf16_t)f0.z;  a[kt][3]  = (bf16_t)f0.w;
        a[kt][4]  = (bf16_t)f1.x;  a[kt][5]  = (bf16_t)f1.y;
        a[kt][6]  = (bf16_t)f1.z;  a[kt][7]  = (bf16_t)f1.w;
        a[kt][8]  = (bf16_t)f2.x;  a[kt][9]  = (bf16_t)f2.y;
        a[kt][10] = (bf16_t)f2.z;  a[kt][11] = (bf16_t)f2.w;
        a[kt][12] = (bf16_t)f3.x;  a[kt][13] = (bf16_t)f3.y;
        a[kt][14] = (bf16_t)f3.z;  a[kt][15] = (bf16_t)f3.w;
    }

#pragma unroll
    for (int nt = 0; nt < NT; ++nt) {
        v8f acc = {};
#pragma unroll
        for (int kt = 0; kt < KT; ++kt) {
            // B 32x16 bf16: lane holds col n = nt*16+lo;
            //   element j   -> K = kt*32 + 8*half + j       (j = 0..7)
            //   element 8+j -> K = kt*32 + 16 + 8*half + j
            // With Wt[n][k] that is two 16B chunks (16B-aligned).
            const bf16_t* bp = Wt + (size_t)(nt * 16 + lo) * K + kt * 32 + 8 * half;
            typedef bf16_t v8bf __attribute__((ext_vector_type(8)));
            v8bf b0 = *(const v8bf*)(bp);
            v8bf b1 = *(const v8bf*)(bp + 16);
            v16bf b = __builtin_shufflevector(b0, b1, 0, 1, 2, 3, 4, 5, 6, 7, 8, 9, 10, 11, 12, 13, 14, 15);
            acc = __builtin_amdgcn_wmma_f32_16x16x32_bf16(
                /*neg_a=*/false, a[kt], /*neg_b=*/false, b,
                /*c_mod=*/(short)0, acc, /*reuse_a=*/false, /*reuse_b=*/false);
        }
        // C/D 16x16 f32: VGPR r -> row m0 + r + 8*half, col nt*16 + lo
#pragma unroll
        for (int r = 0; r < 8; ++r)
            Out[(size_t)(m0 + r + 8 * half) * NC + nt * 16 + lo] = acc[r];
    }
}

// ---------------- edge aggregation ------------------------------------------
// One wave (32 lanes) per edge; each lane moves 4 consecutive floats of the
// 128-wide feature row: coalesced 512B gather + 4 f32 atomics per lane.

__global__ void aggregate_edges(const float* __restrict__ t,
                                const int* __restrict__ src,
                                const int* __restrict__ dst,
                                const float* __restrict__ dinv,
                                float* __restrict__ acc, int E) {
    long long gid = (long long)blockIdx.x * blockDim.x + threadIdx.x;
    int e = (int)(gid >> 5);
    if (e >= E) return;
    int l = (int)(gid & 31);
    int s = src[e];
    int d = dst[e];
    float c = dinv[s] * dinv[d];
    float4 v = ((const float4*)(t + (size_t)s * 128))[l];
    float* out = acc + (size_t)d * 128 + l * 4;
    atomicAdd(out + 0, v.x * c);
    atomicAdd(out + 1, v.y * c);
    atomicAdd(out + 2, v.z * c);
    atomicAdd(out + 3, v.w * c);
}

// Epilogue: add self-loop message (t[i]*dinv[i]^2), bias, ReLU; float4-wide.
__global__ void bias_relu_self(float* __restrict__ h, const float* __restrict__ t,
                               const float* __restrict__ dinv,
                               const float* __restrict__ bias, int n) {
    int i = blockIdx.x * blockDim.x + threadIdx.x;   // over n*32 float4s
    if (i >= n * 32) return;
    int node = i >> 5;
    int f4 = i & 31;
    float di = dinv[node];
    float c = di * di;
    float4 hv = ((const float4*)h)[i];
    float4 tv = ((const float4*)t)[i];
    float4 bv = ((const float4*)bias)[f4];
    float4 r;
    r.x = fmaxf(fmaf(tv.x, c, hv.x) + bv.x, 0.0f);
    r.y = fmaxf(fmaf(tv.y, c, hv.y) + bv.y, 0.0f);
    r.z = fmaxf(fmaf(tv.z, c, hv.z) + bv.z, 0.0f);
    r.w = fmaxf(fmaf(tv.w, c, hv.w) + bv.w, 0.0f);
    ((float4*)h)[i] = r;
}

// ---------------- pooling + readout -----------------------------------------

__global__ void pool_sum(const float* __restrict__ h, const int* __restrict__ batch,
                         float* __restrict__ sums, float* __restrict__ cnt, int n) {
    long long gid = (long long)blockIdx.x * blockDim.x + threadIdx.x;
    int node = (int)(gid >> 5);
    if (node >= n) return;
    int l = (int)(gid & 31);
    int g = batch[node];
    float4 v = ((const float4*)(h + (size_t)node * 128))[l];
    float* out = sums + (size_t)g * 128 + l * 4;
    atomicAdd(out + 0, v.x);
    atomicAdd(out + 1, v.y);
    atomicAdd(out + 2, v.z);
    atomicAdd(out + 3, v.w);
    if (l == 0) atomicAdd(&cnt[g], 1.0f);
}

__global__ void final_linear(const float* __restrict__ sums, const float* __restrict__ cnt,
                             const float* __restrict__ Wlin, const float* __restrict__ blin,
                             float* __restrict__ out) {
    int g = blockIdx.x;      // G blocks
    int o = threadIdx.x;     // OUT threads
    float c = cnt[g];
    c = c > 1.0f ? c : 1.0f;
    float inv = 1.0f / c;
    float a = blin[o];
#pragma unroll 8
    for (int f = 0; f < 128; ++f) a += sums[g * 128 + f] * inv * Wlin[f * 32 + o];
    out[g * 32 + o] = a;
}

// ---------------- host orchestration ----------------------------------------

extern "C" void kernel_launch(void* const* d_in, const int* in_sizes, int n_in,
                              void* d_out, int out_size, void* d_ws, size_t ws_size,
                              hipStream_t stream) {
    const float* x    = (const float*)d_in[0];
    const int*   edge = (const int*)d_in[1];     // [2, E] flat
    const int*   batch= (const int*)d_in[2];
    const float* W1   = (const float*)d_in[3];
    const float* b1   = (const float*)d_in[4];
    const float* W2   = (const float*)d_in[5];
    const float* b2   = (const float*)d_in[6];
    const float* Wlin = (const float*)d_in[7];
    const float* blin = (const float*)d_in[8];
    float* out = (float*)d_out;

    const int FIN = 64, H = 128, OUTF = 32;
    const int n = in_sizes[0] / FIN;        // 100000
    const int E = in_sizes[1] / 2;          // 1600000
    const int G = out_size / OUTF;          // 64
    const int* src = edge;
    const int* dst = edge + E;

    // workspace (floats): deg[n] | dinv[n] | t[n*H] | h[n*H] | sums[G*H] | cnt[G] | Wt(bf16)
    float* ws   = (float*)d_ws;
    float* deg  = ws;
    float* dinv = deg + n;
    float* t    = dinv + n;
    float* h    = t + (size_t)n * H;
    float* sums = h + (size_t)n * H;
    float* cnt  = sums + (size_t)G * H;
    uintptr_t wtp = (uintptr_t)(cnt + G);
    wtp = (wtp + 255) & ~(uintptr_t)255;
    bf16_t* Wt = (bf16_t*)wtp;              // up to H*H bf16 = 32KB, reused per layer

    const int TPB = 256;
    const size_t nh = (size_t)n * H;
    dim3 blkNH4((unsigned)((nh / 4 + TPB - 1) / TPB));
    dim3 blkE((unsigned)(((size_t)E + TPB - 1) / TPB));
    dim3 blkE32((unsigned)(((size_t)E * 32 + TPB - 1) / TPB));
    dim3 blkN32((unsigned)(((size_t)n * 32 + TPB - 1) / TPB));
    dim3 blkN((unsigned)((n + TPB - 1) / TPB));
    dim3 blkN4((unsigned)((n / 4 + TPB - 1) / TPB));
    unsigned gemmBlocks = (unsigned)((n / 16 + 7) / 8);

    // degrees (self-loop contributes 1) and d^{-1/2}
    fill4_kernel<<<blkN4, TPB, 0, stream>>>((float4*)deg, (size_t)n / 4, 1.0f);
    deg_kernel<<<blkE, TPB, 0, stream>>>(dst, deg, E);
    dinv_kernel<<<blkN, TPB, 0, stream>>>(deg, dinv, n);

    // ---- layer 1: t = x @ W1; h = relu(scatter(t) + selfloop + b1)
    wcvt_kernel<<<dim3((unsigned)((FIN * H + TPB - 1) / TPB)), TPB, 0, stream>>>(W1, Wt, FIN, H);
    gemm_bf16_wmma<64, 128><<<gemmBlocks, TPB, 0, stream>>>(x, Wt, t, n);
    fill4_kernel<<<blkNH4, TPB, 0, stream>>>((float4*)h, nh / 4, 0.0f);
    aggregate_edges<<<blkE32, TPB, 0, stream>>>(t, src, dst, dinv, h, E);
    bias_relu_self<<<blkN32, TPB, 0, stream>>>(h, t, dinv, b1, n);

    // ---- layer 2: t = h @ W2; h = relu(scatter(t) + selfloop + b2)
    wcvt_kernel<<<dim3((unsigned)((H * H + TPB - 1) / TPB)), TPB, 0, stream>>>(W2, Wt, H, H);
    gemm_bf16_wmma<128, 128><<<gemmBlocks, TPB, 0, stream>>>(h, Wt, t, n);
    fill4_kernel<<<blkNH4, TPB, 0, stream>>>((float4*)h, nh / 4, 0.0f);
    aggregate_edges<<<blkE32, TPB, 0, stream>>>(t, src, dst, dinv, h, E);
    bias_relu_self<<<blkN32, TPB, 0, stream>>>(h, t, dinv, b2, n);

    // ---- global mean pool + linear readout
    fill4_kernel<<<dim3((unsigned)((((size_t)G * H + G) / 4 + TPB - 1) / TPB)), TPB, 0, stream>>>(
        (float4*)sums, ((size_t)G * H + G) / 4, 0.0f);
    pool_sum<<<blkN32, TPB, 0, stream>>>(h, batch, sums, cnt, n);
    final_linear<<<dim3((unsigned)G), OUTF, 0, stream>>>(sums, cnt, Wlin, blin, out);
}
